// RNNLayer_50989851738663
// MI455X (gfx1250) — matrix operand order repaired and checked by reading, not compiled, not run on hardware
//
#include <hip/hip_runtime.h>

typedef __bf16 bf16;
typedef __attribute__((ext_vector_type(8)))  __bf16 v8bf;
typedef __attribute__((ext_vector_type(16))) __bf16 v16bf;
typedef __attribute__((ext_vector_type(8)))  float   v8f;

// explicit global-address-space access (forces global_load/global_store, not flat)
#define AS1C(T, p) ((const __attribute__((address_space(1))) T*)(unsigned long long)(p))
#define AS1M(T, p) ((__attribute__((address_space(1))) T*)(unsigned long long)(p))

constexpr int SEQ = 8192;
constexpr int EMB = 2048;
constexpr int HID = 2048;
constexpr int CH  = 32;          // chunk length (U^32 boundary hop)
constexpr int NC  = SEQ / CH;    // 256 chunks

// epilogue flags (template parameters -> dead-code-free epilogues)
constexpr int WF32    = 1;
constexpr int WBF16   = 2;
constexpr int LCIN    = 4;
constexpr int ADDRELU = 8;

// ---------------- helper kernels ----------------

__global__ void k_f32_to_bf16(const float* __restrict__ in, bf16* __restrict__ out, int n) {
    int i = blockIdx.x * blockDim.x + threadIdx.x;
    if (i < n) *AS1M(bf16, out + i) = (bf16)*AS1C(float, in + i);
}

__global__ void k_fill_zero_bf16(bf16* __restrict__ p, int n) {
    int i = blockIdx.x * blockDim.x + threadIdx.x;
    if (i < n) *AS1M(bf16, p + i) = (bf16)0.0f;
}

// out[c][r] = in[r][c] ; in is R x C row-major
__global__ void k_transpose_bf16(const bf16* __restrict__ in, bf16* __restrict__ out, int R, int C) {
    int i = blockIdx.x * blockDim.x + threadIdx.x;
    if (i < R * C) {
        int r = i / C, c = i % C;
        *AS1M(bf16, out + (size_t)c * R + r) = *AS1C(bf16, in + i);
    }
}

__global__ void k_init_g(const float* __restrict__ h0, float* __restrict__ g0, bf16* __restrict__ gb0) {
    int i = blockIdx.x * blockDim.x + threadIdx.x;
    if (i < HID) {
        float v = *AS1C(float, h0 + i);
        *AS1M(float, g0 + i) = v;
        *AS1M(bf16, gb0 + i) = (bf16)v;
    }
}

// ---------------- register-blocked WMMA GEMM ----------------
// D[M x N] = A[M x K](bf16 row-major, lda) * B[K x N](bf16 COLUMN-contiguous, ldb = col stride)
// Each wave computes an MT x NT grid of 16x16 tiles: each A fragment is reused NT
// times and each B fragment MT times per K-step (fragment traffic / FLOP drops 2*MT*NT/(MT+NT)-fold).
template<int MT, int NT, int FLAGS>
__global__ __launch_bounds__(256)
void k_gemm_wmma(const bf16* __restrict__ A, long lda,
                 const bf16* __restrict__ B, long ldb, int K,
                 int MtilesW, int NtilesW,   // counts of (16*MT) x (16*NT) wave-tiles
                 const float* __restrict__ Cin, long cSM, long cSN,
                 float* __restrict__ Df, long dSM, long dSN,
                 bf16* __restrict__ Db, long bSM, long bSN)
{
    const int lane = threadIdx.x & 31;
    const int wv   = threadIdx.x >> 5;
    const int idx  = blockIdx.x * 8 + wv;            // one wave-tile per wave
    if (idx >= MtilesW * NtilesW) return;            // wave-uniform: EXEC stays all-ones
    const int m0 = (idx % MtilesW) * (16 * MT);
    const int n0 = (idx / MtilesW) * (16 * NT);
    const int r  = lane & 15;
    const int hi = lane >> 4;                        // 0 for lanes 0-15, 1 for 16-31

    v8f acc[MT][NT];
#pragma unroll
    for (int mi = 0; mi < MT; ++mi)
#pragma unroll
        for (int ni = 0; ni < NT; ++ni)
#pragma unroll
            for (int v = 0; v < 8; ++v) acc[mi][ni][v] = 0.0f;

    if constexpr (FLAGS & LCIN) {
#pragma unroll
        for (int mi = 0; mi < MT; ++mi)
#pragma unroll
            for (int ni = 0; ni < NT; ++ni)
#pragma unroll
                for (int v = 0; v < 8; ++v) {
                    long m = m0 + mi * 16 + hi * 8 + v;
                    long n = n0 + ni * 16 + r;
                    acc[mi][ni][v] = *AS1C(float, Cin + m * cSM + n * cSN);
                }
    }

    // A fragment (ISA 16-bit A 16x32): lane r holds row; two 8-elt contiguous runs:
    //   lanes 0-15: K = [0..7],[16..23];  lanes 16-31: K = [8..15],[24..31]
    const bf16* ap[MT];
#pragma unroll
    for (int mi = 0; mi < MT; ++mi)
        ap[mi] = A + (long)(m0 + mi * 16 + r) * lda + hi * 8;
    // B fragment (16-bit B 32x16): lane r holds col; two 8-elt runs matching A:
    //   lanes 0-15: K = [0..7],[16..23];  lanes 16-31: K = [8..15],[24..31]
    const bf16* bp[NT];
#pragma unroll
    for (int ni = 0; ni < NT; ++ni)
        bp[ni] = B + (long)(n0 + ni * 16 + r) * ldb + hi * 8;

    for (int k0 = 0; k0 < K; k0 += 32) {
        v16bf af[MT], bfr[NT];
#pragma unroll
        for (int mi = 0; mi < MT; ++mi) {
            v8bf a0 = *AS1C(v8bf, ap[mi]);
            v8bf a1 = *AS1C(v8bf, ap[mi] + 16);
            af[mi] = __builtin_shufflevector(a0, a1, 0,1,2,3,4,5,6,7,8,9,10,11,12,13,14,15);
            __builtin_prefetch(ap[mi] + 256, 0, 3);   // prefetch 8 K-steps ahead into GL2
            ap[mi] += 32;
        }
#pragma unroll
        for (int ni = 0; ni < NT; ++ni) {
            v8bf b0 = *AS1C(v8bf, bp[ni]);
            v8bf b1 = *AS1C(v8bf, bp[ni] + 16);
            bfr[ni] = __builtin_shufflevector(b0, b1, 0,1,2,3,4,5,6,7,8,9,10,11,12,13,14,15);
            __builtin_prefetch(bp[ni] + 256, 0, 3);
            bp[ni] += 32;
        }
#pragma unroll
        for (int mi = 0; mi < MT; ++mi)
#pragma unroll
            for (int ni = 0; ni < NT; ++ni)
                acc[mi][ni] = __builtin_amdgcn_wmma_f32_16x16x32_bf16(
                    false, af[mi], false, bfr[ni], (short)0, acc[mi][ni], false, false);
    }

    // Epilogue. C/D layout: lane holds col (lane%16); VGPR v holds row 8*hi + v
#pragma unroll
    for (int mi = 0; mi < MT; ++mi)
#pragma unroll
        for (int ni = 0; ni < NT; ++ni)
#pragma unroll
            for (int v = 0; v < 8; ++v) {
                long m = m0 + mi * 16 + hi * 8 + v;
                long n = n0 + ni * 16 + r;
                float o = acc[mi][ni][v];
                if constexpr (FLAGS & WBF16)
                    *AS1M(bf16, Db + m * bSM + n * bSN) = (bf16)o;   // pre-relu copy
                if constexpr (FLAGS & ADDRELU) {
                    auto dp = AS1M(float, Df + m * dSM + n * dSN);
                    float t = *dp + o;
                    *dp = t > 0.0f ? t : 0.0f;
                } else if constexpr (FLAGS & WF32) {
                    *AS1M(float, Df + m * dSM + n * dSN) = o;
                }
            }
}

// ---------------- boundary matvec: g_out = V * g + radd ----------------
// one wave (32 lanes) per output row; V is bf16 row-major HIDxHID (L2-resident, 8 MB)
__global__ __launch_bounds__(256)
void k_matvec_step(const bf16* __restrict__ V, const float* __restrict__ g,
                   const float* __restrict__ radd,
                   float* __restrict__ gout, bf16* __restrict__ gb_out)
{
    const int lane = threadIdx.x & 31;
    const int row  = blockIdx.x * (blockDim.x >> 5) + (threadIdx.x >> 5);
    const bf16* vr = V + (long)row * HID;
    float s = 0.0f;
    for (int k = lane; k < HID; k += 32)
        s += (float)*AS1C(bf16, vr + k) * *AS1C(float, g + k);
#pragma unroll
    for (int off = 16; off; off >>= 1) s += __shfl_xor(s, off, 32);
    if (lane == 0) {
        float y = s + *AS1C(float, radd + row);
        *AS1M(float, gout + row) = y;
        if (gb_out) *AS1M(bf16, gb_out + row) = (bf16)y;
    }
}

// ---------------- host orchestration ----------------

extern "C" void kernel_launch(void* const* d_in, const int* in_sizes, int n_in,
                              void* d_out, int out_size, void* d_ws, size_t ws_size,
                              hipStream_t stream)
{
    const float* X  = (const float*)d_in[0];   // [SEQ, EMB]
    const float* h0 = (const float*)d_in[1];   // [HID]
    const float* W  = (const float*)d_in[2];   // [HID, EMB]
    const float* U  = (const float*)d_in[3];   // [HID, HID]
    float* OUT = (float*)d_out;                // [SEQ, HID]

    char* ws = (char*)d_ws;
    size_t off = 0;
    auto alloc = [&](size_t bytes) -> void* {
        void* p = ws + off;
        off += (bytes + 255) & ~(size_t)255;
        return p;
    };
    bf16* Xbf  = (bf16*)alloc((size_t)SEQ * EMB * 2);   // 32 MB
    bf16* Wbf  = (bf16*)alloc((size_t)HID * EMB * 2);   //  8 MB
    bf16* Ubf  = (bf16*)alloc((size_t)HID * HID * 2);   //  8 MB
    bf16* powT = (bf16*)alloc((size_t)HID * HID * 2);   //  8 MB (transposed operand)
    bf16* powA = (bf16*)alloc((size_t)HID * HID * 2);   //  8 MB
    bf16* powB = (bf16*)alloc((size_t)HID * HID * 2);   //  8 MB
    bf16* L0   = (bf16*)alloc((size_t)NC * HID * 2);    //  1 MB  local-scan state ping
    bf16* L1   = (bf16*)alloc((size_t)NC * HID * 2);    //  1 MB  local-scan state pong
    bf16* G0   = (bf16*)alloc((size_t)NC * HID * 2);    //  1 MB  boundary states (bf16)
    bf16* G1   = (bf16*)alloc((size_t)NC * HID * 2);    //  1 MB
    float* gbuf = (float*)alloc((size_t)(NC + 1) * HID * 4); // 2 MB fp32 boundary chain

    const int T256 = 256;

    // 1) fp32 -> bf16 operand conversions
    k_f32_to_bf16<<<(SEQ * EMB + T256 - 1) / T256, T256, 0, stream>>>(X, Xbf, SEQ * EMB);
    k_f32_to_bf16<<<(HID * EMB + T256 - 1) / T256, T256, 0, stream>>>(W, Wbf, HID * EMB);
    k_f32_to_bf16<<<(HID * HID + T256 - 1) / T256, T256, 0, stream>>>(U, Ubf, HID * HID);

    // 2) B = X * W^T  (fp32, straight into d_out). W row-major == W^T column-contiguous.
    {
        int MtW = SEQ / 64, NtW = HID / 64;                 // 4x4 blocking: 128 x 32 wave-tiles
        k_gemm_wmma<4, 4, WF32><<<(MtW * NtW) / 8, T256, 0, stream>>>(
            Xbf, EMB, Wbf, EMB, EMB, MtW, NtW,
            nullptr, 0, 0,
            OUT, (long)HID, 1,
            nullptr, 0, 0);
    }

    // 3) V = U^32 by 5 squarings (bf16 out, transpose materialized per step)
    {
        bf16* P = Ubf;
        bf16* dsts[5] = { powA, powB, powA, powB, powA };
        int MtW = HID / 64, NtW = HID / 64;                 // 32 x 32 wave-tiles
        for (int s = 0; s < 5; ++s) {
            k_transpose_bf16<<<(HID * HID + T256 - 1) / T256, T256, 0, stream>>>(P, powT, HID, HID);
            k_gemm_wmma<4, 4, WBF16><<<(MtW * NtW) / 8, T256, 0, stream>>>(
                P, HID, powT, HID, HID, MtW, NtW,
                nullptr, 0, 0,
                nullptr, 0, 0,
                dsts[s], (long)HID, 1);
            P = dsts[s];
        }
        // P == powA == U^32
    }
    bf16* V32 = powA;

    // 4) Pass A: within-chunk scan, batched over all 256 chunks.
    //    L_j = U * L_{j-1} + B_j ; l values overwrite d_out rows in place, bf16 copy kept.
    //    2x2 blocking keeps 512 waves in flight per (serial) step.
    k_fill_zero_bf16<<<(NC * HID + T256 - 1) / T256, T256, 0, stream>>>(L0, NC * HID);
    {
        bf16* Lp = L0; bf16* Lc = L1;
        int MtW = HID / 32, NtW = NC / 32;                  // 64 x 8 wave-tiles
        for (int j = 0; j < CH; ++j) {
            float* rows = OUT + (size_t)j * HID;            // element (m, chunk c) at c*CH*HID + m
            k_gemm_wmma<2, 2, WF32 | WBF16 | LCIN><<<(MtW * NtW) / 8, T256, 0, stream>>>(
                Ubf, HID, Lp, HID, HID, MtW, NtW,
                rows, 1, (long)CH * HID,
                rows, 1, (long)CH * HID,
                Lc, 1, (long)HID);
            bf16* t = Lp; Lp = Lc; Lc = t;
        }
    }

    // 5) Pass B: serial boundary chain g_{c+1} = V * g_c + l_{c,CH-1} (only 255 steps).
    k_init_g<<<(HID + T256 - 1) / T256, T256, 0, stream>>>(h0, gbuf, G0);
    for (int c = 0; c < NC - 1; ++c) {
        k_matvec_step<<<HID / 8, T256, 0, stream>>>(
            V32,
            gbuf + (size_t)c * HID,
            OUT + ((size_t)c * CH + (CH - 1)) * HID,
            gbuf + (size_t)(c + 1) * HID,
            G0 + (size_t)(c + 1) * HID);
    }

    // 6) Pass C: corrections G_j = U * G_{j-1} (G_{-1} = boundary states);
    //    out_row(c*CH+j) = relu(l + G_j[:,c]) fused in epilogue.
    {
        bf16* Gp = G0; bf16* Gc = G1;
        int MtW = HID / 32, NtW = NC / 32;
        for (int j = 0; j < CH; ++j) {
            float* rows = OUT + (size_t)j * HID;
            k_gemm_wmma<2, 2, WBF16 | ADDRELU><<<(MtW * NtW) / 8, T256, 0, stream>>>(
                Ubf, HID, Gp, HID, HID, MtW, NtW,
                nullptr, 0, 0,
                rows, 1, (long)CH * HID,
                Gc, 1, (long)HID);
            bf16* t = Gp; Gp = Gc; Gc = t;
        }
    }

    (void)in_sizes; (void)n_in; (void)out_size; (void)ws_size;
}
